// AlignmentNetwork_40200893890724
// MI455X (gfx1250) — matrix operand order repaired and checked by reading, not compiled, not run on hardware
//
#include <hip/hip_runtime.h>
#include <hip/hip_bf16.h>

// ---------------- problem constants ----------------
#define BB    32
#define TEN   512
#define TDE   2048
#define CK    512
#define CQ    80
#define CQP   96      // CQ padded to multiple of 32 for WMMA K-loop
#define CA    128
#define CH1K  1024    // 2*CK
#define CH1Q  160     // 2*CQ
#define TEMPC 0.0005f
#define NEG_INF (-__builtin_inff())

typedef __attribute__((ext_vector_type(16))) _Float16 v16h;
typedef __attribute__((ext_vector_type(8)))  _Float16 h8v;
typedef __attribute__((ext_vector_type(8)))  float    v8f;

// ---------------- utility kernels ----------------
__global__ void zero_f32(float* __restrict__ p, size_t n) {
    size_t i = (size_t)blockIdx.x * blockDim.x + threadIdx.x;
    size_t st = (size_t)gridDim.x * blockDim.x;
    for (; i < n; i += st) p[i] = 0.0f;
}

// f32 [b][T][C] -> f16 [b][Tld][Cp] at row offset halo, zero-padded channels.
// (halo rows themselves are zeroed separately.)
__global__ void cvt_act(const float* __restrict__ src, _Float16* __restrict__ dst,
                        int batches, int T, int C, int Cp, int Tld, int halo) {
    size_t n = (size_t)batches * T * Cp;
    size_t i = (size_t)blockIdx.x * blockDim.x + threadIdx.x;
    size_t st = (size_t)gridDim.x * blockDim.x;
    for (; i < n; i += st) {
        int c = (int)(i % Cp);
        size_t rt = i / Cp;
        int t = (int)(rt % T);
        int b = (int)(rt / T);
        dst[((size_t)b * Tld + halo + t) * Cp + c] =
            (c < C) ? (_Float16)src[((size_t)b * T + t) * C + c] : (_Float16)0.0f;
    }
}

// f32 weights [Co][Ci][taps] -> f16 [tap][Co][Cp] (row-major K, zero-padded)
__global__ void cvt_w(const float* __restrict__ src, _Float16* __restrict__ dst,
                      int Co, int Ci, int taps, int Cp) {
    size_t n = (size_t)taps * Co * Cp;
    size_t i = (size_t)blockIdx.x * blockDim.x + threadIdx.x;
    size_t st = (size_t)gridDim.x * blockDim.x;
    for (; i < n; i += st) {
        int cp  = (int)(i % Cp);
        int co  = (int)((i / Cp) % Co);
        int tap = (int)(i / ((size_t)Cp * Co));
        dst[i] = (cp < Ci) ? (_Float16)src[((size_t)co * Ci + cp) * taps + tap]
                           : (_Float16)0.0f;
    }
}

// ---------------- WMMA conv-as-GEMM ----------------
// Out[b][t][co] (fp16, row stride Mld) = act( sum_tap W[tap] x In[b][halo+t+tap-taps/2][*] + bias )
// One wave computes a 16(M=co) x 32(N=t) tile: 1 A frag, 2 B frags, 2 accumulators.
// Input buffers for taps==3 carry zeroed halo rows so all loads are unconditional.
__global__ void conv_gemm_wmma(const _Float16* __restrict__ W,    // [taps][M][K]
                               const _Float16* __restrict__ act,  // [B][Tld][K]
                               const float*    __restrict__ bias, // [M]
                               _Float16*       __restrict__ out,  // [B][T][Mld]
                               int M, int K, int T, int taps, int Mld, int doRelu,
                               int Tld, int halo) {
    const int b      = blockIdx.y;
    const int tilesM = M >> 4;
    const int tm     = (blockIdx.x % tilesM) << 4;
    const int tn     = (blockIdx.x / tilesM) << 5;   // 32-wide N tile
    const int lane   = threadIdx.x;
    const int hs     = lane >> 4;      // half-wave select
    const int l15    = lane & 15;

    const _Float16* actB = act + ((size_t)b * Tld + halo) * K;
    const int n0   = tn + l15;         // output time index, first D tile
    const int n1   = tn + 16 + l15;    // second D tile
    const int arow = tm + l15;         // A row (co)

    v8f acc0 = {}, acc1 = {};
    for (int tap = 0; tap < taps; ++tap) {
        const _Float16* Wt = W + (size_t)tap * M * K;
        const int sh = tap - (taps >> 1);
        const _Float16* brow0 = actB + (size_t)(n0 + sh) * K;
        const _Float16* brow1 = actB + (size_t)(n1 + sh) * K;
        for (int k0 = 0; k0 < K; k0 += 32) {
            v16h a;
            {   // A frag: row=arow, K = k0 + hs*16 + {0..15} (contiguous)
                const _Float16* ap = Wt + (size_t)arow * K + k0 + (hs << 4);
                *((h8v*)&a)       = *((const h8v*)ap);
                *(((h8v*)&a) + 1) = *((const h8v*)(ap + 8));
            }
            v16h b0, b1;
            {   // B frags: col=n, K = k0 + hs*16 + {0..15} (contiguous)
                const _Float16* bp0 = brow0 + k0 + (hs << 4);
                *((h8v*)&b0)       = *((const h8v*)bp0);
                *(((h8v*)&b0) + 1) = *((const h8v*)(bp0 + 8));
                const _Float16* bp1 = brow1 + k0 + (hs << 4);
                *((h8v*)&b1)       = *((const h8v*)bp1);
                *(((h8v*)&b1) + 1) = *((const h8v*)(bp1 + 8));
            }
            acc0 = __builtin_amdgcn_wmma_f32_16x16x32_f16(
                false, a, false, b0, (short)0, acc0, false, false);
            acc1 = __builtin_amdgcn_wmma_f32_16x16x32_f16(
                false, a, false, b1, (short)0, acc1, false, false);
        }
    }
    // epilogue: bias + relu + fp16 pack; contiguous 8-half stores into [t][co]
    const int mbase = hs << 3;
    h8v o0, o1;
#pragma unroll
    for (int r = 0; r < 8; ++r) {
        float bv = bias[tm + mbase + r];
        float v0 = acc0[r] + bv;
        float v1 = acc1[r] + bv;
        if (doRelu) { v0 = v0 > 0.0f ? v0 : 0.0f; v1 = v1 > 0.0f ? v1 : 0.0f; }
        o0[r] = (_Float16)v0;
        o1[r] = (_Float16)v1;
    }
    *((h8v*)(out + ((size_t)b * T + n0) * Mld + tm + mbase)) = o0;
    *((h8v*)(out + ((size_t)b * T + n1) * Mld + tm + mbase)) = o1;
}

// ---------------- squared-norm rows ----------------
__global__ void norms_kernel(const _Float16* __restrict__ h, float* __restrict__ out,
                             size_t rows, int C) {
    size_t r = (size_t)blockIdx.x * blockDim.x + threadIdx.x;
    if (r >= rows) return;
    const _Float16* p = h + r * C;
    float s = 0.0f;
    for (int c = 0; c < C; ++c) { float v = (float)p[c]; s += v * v; }
    out[r] = s;
}

// ---------------- score GEMM: logits = -TEMP*(|q|^2+|k|^2-2 q.k) ----------------
// One wave computes a 16(t) x 32(s) tile.
__global__ void score_gemm_wmma(const _Float16* __restrict__ qh,  // [B][TDE][CA]
                                const _Float16* __restrict__ kh,  // [B][TEN][CA]
                                const float* __restrict__ q2n,    // [B][TDE]
                                const float* __restrict__ k2n,    // [B][TEN]
                                float* __restrict__ logits) {     // [B][TDE][TEN]
    const int b      = blockIdx.y;
    const int tilesN = TEN >> 5;                      // 32-wide s tiles
    const int tm     = (blockIdx.x / tilesN) << 4;    // t tile
    const int tn     = (blockIdx.x % tilesN) << 5;    // s tile
    const int lane   = threadIdx.x;
    const int hs     = lane >> 4;
    const int l15    = lane & 15;

    const _Float16* qb = qh + (size_t)b * TDE * CA;
    const _Float16* kb = kh + (size_t)b * TEN * CA;
    const int arow = tm + l15;        // t
    const int n0   = tn + l15;        // s, first tile
    const int n1   = tn + 16 + l15;   // s, second tile

    v8f acc0 = {}, acc1 = {};
#pragma unroll
    for (int k0 = 0; k0 < CA; k0 += 32) {
        v16h a;
        {
            const _Float16* ap = qb + (size_t)arow * CA + k0 + (hs << 4);
            *((h8v*)&a)       = *((const h8v*)ap);
            *(((h8v*)&a) + 1) = *((const h8v*)(ap + 8));
        }
        v16h b0, b1;
        {
            const _Float16* bp0 = kb + (size_t)n0 * CA + k0 + (hs << 4);
            *((h8v*)&b0)       = *((const h8v*)bp0);
            *(((h8v*)&b0) + 1) = *((const h8v*)(bp0 + 8));
            const _Float16* bp1 = kb + (size_t)n1 * CA + k0 + (hs << 4);
            *((h8v*)&b1)       = *((const h8v*)bp1);
            *(((h8v*)&b1) + 1) = *((const h8v*)(bp1 + 8));
        }
        acc0 = __builtin_amdgcn_wmma_f32_16x16x32_f16(
            false, a, false, b0, (short)0, acc0, false, false);
        acc1 = __builtin_amdgcn_wmma_f32_16x16x32_f16(
            false, a, false, b1, (short)0, acc1, false, false);
    }
    const float* q2b = q2n + (size_t)b * TDE;
    const float  k2v0 = k2n[(size_t)b * TEN + n0];
    const float  k2v1 = k2n[(size_t)b * TEN + n1];
    float* lp = logits + (size_t)b * TDE * TEN;
    const int mbase = hs << 3;
#pragma unroll
    for (int r = 0; r < 8; ++r) {
        int t = tm + mbase + r;
        float q2v = q2b[t];
        lp[(size_t)t * TEN + n0] = -TEMPC * (q2v + k2v0 - 2.0f * acc0[r]);
        lp[(size_t)t * TEN + n1] = -TEMPC * (q2v + k2v1 - 2.0f * acc1[r]);
    }
}

// ---------------- prior transpose [B][TEN][TDE] -> [B][TDE][TEN] ----------------
__global__ void transpose_prior(const float* __restrict__ in, float* __restrict__ out) {
    __shared__ float tile[32][33];
    const int b  = blockIdx.z;
    const int t0 = blockIdx.x * 32;   // TDE base
    const int s0 = blockIdx.y * 32;   // TEN base
    const int x  = threadIdx.x;
    const float* ib = in  + (size_t)b * TEN * TDE;
    float*       ob = out + (size_t)b * TDE * TEN;
    for (int j = threadIdx.y; j < 32; j += 8)
        tile[j][x] = ib[(size_t)(s0 + j) * TDE + t0 + x];
    __syncthreads();
    for (int j = threadIdx.y; j < 32; j += 8)
        ob[(size_t)(t0 + j) * TEN + s0 + x] = tile[x][j];
}

// ---------------- fused double softmax row kernel ----------------
// alp holds raw logits on entry; on exit: attn_logp. attn gets softmax(attn_logp).
__global__ void softmax_kernel(float* __restrict__ alp, float* __restrict__ attn,
                               const float* __restrict__ priorT) {
    const size_t row = blockIdx.x;               // b*TDE + t
    float* rp       = alp  + row * TEN;
    float* ap       = attn + row * TEN;
    const float* pp = priorT + row * TEN;
    __shared__ float red[256];
    const int tid = threadIdx.x;

    float x0 = rp[tid], x1 = rp[tid + 256];
    // pass 1: logsumexp over s
    float m = fmaxf(x0, x1);
    red[tid] = m; __syncthreads();
    for (int o = 128; o > 0; o >>= 1) { if (tid < o) red[tid] = fmaxf(red[tid], red[tid + o]); __syncthreads(); }
    m = red[0]; __syncthreads();
    float e = __expf(x0 - m) + __expf(x1 - m);
    red[tid] = e; __syncthreads();
    for (int o = 128; o > 0; o >>= 1) { if (tid < o) red[tid] += red[tid + o]; __syncthreads(); }
    float lse = m + __logf(red[0]); __syncthreads();

    float a0 = x0 - lse + __logf(pp[tid] + 1e-8f);
    float a1 = x1 - lse + __logf(pp[tid + 256] + 1e-8f);

    // pass 2: softmax of attn_logp
    float m2 = fmaxf(a0, a1);
    red[tid] = m2; __syncthreads();
    for (int o = 128; o > 0; o >>= 1) { if (tid < o) red[tid] = fmaxf(red[tid], red[tid + o]); __syncthreads(); }
    m2 = red[0]; __syncthreads();
    float e0 = __expf(a0 - m2), e1 = __expf(a1 - m2);
    red[tid] = e0 + e1; __syncthreads();
    for (int o = 128; o > 0; o >>= 1) { if (tid < o) red[tid] += red[tid + o]; __syncthreads(); }
    float inv = 1.0f / red[0];

    rp[tid] = a0; rp[tid + 256] = a1;
    ap[tid] = e0 * inv; ap[tid + 256] = e1 * inv;
}

// ---------------- MAS forward DP (one workgroup per batch) ----------------
__global__ void mas_fwd(const float* __restrict__ attn, unsigned* __restrict__ dirs) {
    const int b = blockIdx.x;
    const int s = threadIdx.x;                    // 0..511
    __shared__ float v[TEN];
    v[s] = 0.0f;
    __syncthreads();
    const float* ab = attn + (size_t)b * TDE * TEN;
    unsigned*    db = dirs + (size_t)b * TDE * (TEN / 32);
    for (int t = 0; t < TDE; ++t) {
        float vj = ab[(size_t)t * TEN + s];
        float vs = v[s];
        float v0 = (s == 0) ? NEG_INF : v[s - 1];
        bool  mm = vs >= v0;
        float vmax = mm ? vs : v0;
        float vnew = (s <= t) ? vmax + vj : NEG_INF;
        unsigned bits = __builtin_amdgcn_ballot_w32(mm);
        if ((s & 31) == 0) db[(size_t)t * (TEN / 32) + (s >> 5)] = bits;
        __syncthreads();
        v[s] = vnew;
        __syncthreads();
    }
}

// ---------------- MAS backtrack + durations ----------------
__global__ void mas_bwd(const unsigned* __restrict__ dirs, float* __restrict__ mas,
                        int* __restrict__ dur) {
    const int b   = blockIdx.x;
    const int tid = threadIdx.x;                  // 0..511
    __shared__ unsigned short idxs[TDE];
    __shared__ int hist[TEN];
    hist[tid] = 0;
    if (tid == 0) {
        int idx = TEN - 1;
        const unsigned* db = dirs + (size_t)b * TDE * (TEN / 32);
        for (int t = TDE - 1; t >= 0; --t) {
            idxs[t] = (unsigned short)idx;
            unsigned w = db[(size_t)t * (TEN / 32) + (idx >> 5)];
            int d = (int)((w >> (idx & 31)) & 1u);
            idx += d - 1;
            if (idx < 0) idx = 0;
            if (idx > TEN - 1) idx = TEN - 1;
        }
    }
    __syncthreads();
    float* mb = mas + (size_t)b * TDE * TEN;
    for (int t = tid; t < TDE; t += blockDim.x) {
        int s = (int)idxs[t];
        mb[(size_t)t * TEN + s] = 1.0f;
        atomicAdd(&hist[s], 1);
    }
    __syncthreads();
    dur[(size_t)b * TEN + tid] = hist[tid];
}

// ---------------- launch ----------------
extern "C" void kernel_launch(void* const* d_in, const int* in_sizes, int n_in,
                              void* d_out, int out_size, void* d_ws, size_t ws_size,
                              hipStream_t stream) {
    const float* x     = (const float*)d_in[0];   // [B, TEN, CK]
    const float* y     = (const float*)d_in[1];   // [B, TDE, CQ]
    const float* prior = (const float*)d_in[4];   // [B, TEN, TDE]
    const float* kw1 = (const float*)d_in[5];
    const float* kb1 = (const float*)d_in[6];
    const float* kw2 = (const float*)d_in[7];
    const float* kb2 = (const float*)d_in[8];
    const float* qw1 = (const float*)d_in[9];
    const float* qb1 = (const float*)d_in[10];
    const float* qw2 = (const float*)d_in[11];
    const float* qb2 = (const float*)d_in[12];
    const float* qw3 = (const float*)d_in[13];
    const float* qb3 = (const float*)d_in[14];

    // workspace carve-out (256B aligned)
    char* ws = (char*)d_ws;
    size_t off = 0;
    auto carve = [&](size_t bytes) -> void* {
        void* p = ws + off; off += (bytes + 255) & ~(size_t)255; return p;
    };
    const int TENH = TEN + 2;   // halo rows for conv-3 inputs
    const int TDEH = TDE + 2;
    _Float16* xh   = (_Float16*)carve((size_t)BB * TENH * CK  * 2);
    _Float16* yh   = (_Float16*)carve((size_t)BB * TDEH * CQP * 2);
    _Float16* k1h  = (_Float16*)carve((size_t)BB * TEN * CH1K * 2);
    _Float16* kh   = (_Float16*)carve((size_t)BB * TEN * CA   * 2);
    _Float16* q1h  = (_Float16*)carve((size_t)BB * TDE * CH1Q * 2);
    _Float16* q2h  = (_Float16*)carve((size_t)BB * TDE * CQP  * 2);
    _Float16* qh   = (_Float16*)carve((size_t)BB * TDE * CA   * 2);
    float*    k2n  = (float*)   carve((size_t)BB * TEN * 4);
    float*    q2n  = (float*)   carve((size_t)BB * TDE * 4);
    float*    priT = (float*)   carve((size_t)BB * TDE * TEN * 4);
    unsigned* dirs = (unsigned*)carve((size_t)BB * TDE * (TEN / 32) * 4);
    _Float16* kw1h = (_Float16*)carve((size_t)3 * CH1K * CK  * 2);
    _Float16* kw2h = (_Float16*)carve((size_t)CA * CH1K      * 2);
    _Float16* qw1h = (_Float16*)carve((size_t)3 * CH1Q * CQP * 2);
    _Float16* qw2h = (_Float16*)carve((size_t)CQ * CH1Q      * 2);
    _Float16* qw3h = (_Float16*)carve((size_t)CA * CQP       * 2);

    // output regions
    const size_t SZ = (size_t)BB * TDE * TEN;
    float* alp  = (float*)d_out;
    float* attn = alp + SZ;
    float* mas  = attn + SZ;
    int*   dur  = (int*)(mas + SZ);

    auto gs = [](size_t n, int bl) { size_t g = (n + bl - 1) / bl;
                                     return (unsigned)(g > 262144 ? 262144 : g); };

    // 0) zero: halo'd conv inputs (guarantees zero halos + pad), q2h pad cols, mas region
    zero_f32<<<gs((size_t)BB * TENH * CK / 2, 256), 256, 0, stream>>>((float*)xh, (size_t)BB * TENH * CK / 2);
    zero_f32<<<gs((size_t)BB * TDEH * CQP / 2, 256), 256, 0, stream>>>((float*)yh, (size_t)BB * TDEH * CQP / 2);
    zero_f32<<<gs((size_t)BB * TDE * CQP / 2, 256), 256, 0, stream>>>((float*)q2h, (size_t)BB * TDE * CQP / 2);
    zero_f32<<<gs(SZ, 256), 256, 0, stream>>>(mas, SZ);

    // 1) fp16 conversions (interior rows only; halos stay zero)
    cvt_act<<<gs((size_t)BB * TEN * CK, 256), 256, 0, stream>>>(x, xh, BB, TEN, CK, CK, TENH, 1);
    cvt_act<<<gs((size_t)BB * TDE * CQP, 256), 256, 0, stream>>>(y, yh, BB, TDE, CQ, CQP, TDEH, 1);
    cvt_w<<<gs((size_t)3 * CH1K * CK, 256), 256, 0, stream>>>(kw1, kw1h, CH1K, CK, 3, CK);
    cvt_w<<<gs((size_t)CA * CH1K, 256), 256, 0, stream>>>(kw2, kw2h, CA, CH1K, 1, CH1K);
    cvt_w<<<gs((size_t)3 * CH1Q * CQP, 256), 256, 0, stream>>>(qw1, qw1h, CH1Q, CQ, 3, CQP);
    cvt_w<<<gs((size_t)CQ * CH1Q, 256), 256, 0, stream>>>(qw2, qw2h, CQ, CH1Q, 1, CH1Q);
    cvt_w<<<gs((size_t)CA * CQP, 256), 256, 0, stream>>>(qw3, qw3h, CA, CQ, 1, CQP);

    // 2) conv stacks as WMMA GEMMs (one wave per 16x32 tile)
    // k path: relu(conv3 512->1024), conv1 1024->128
    conv_gemm_wmma<<<dim3((CH1K/16)*(TEN/32), BB), 32, 0, stream>>>(
        kw1h, xh, kb1, k1h, CH1K, CK, TEN, 3, CH1K, 1, TENH, 1);
    conv_gemm_wmma<<<dim3((CA/16)*(TEN/32), BB), 32, 0, stream>>>(
        kw2h, k1h, kb2, kh, CA, CH1K, TEN, 1, CA, 0, TEN, 0);
    // q path: relu(conv3 80->160), relu(conv1 160->80), conv1 80->128
    conv_gemm_wmma<<<dim3((CH1Q/16)*(TDE/32), BB), 32, 0, stream>>>(
        qw1h, yh, qb1, q1h, CH1Q, CQP, TDE, 3, CH1Q, 1, TDEH, 1);
    conv_gemm_wmma<<<dim3((CQ/16)*(TDE/32), BB), 32, 0, stream>>>(
        qw2h, q1h, qb2, q2h, CQ, CH1Q, TDE, 1, CQP, 1, TDE, 0);
    conv_gemm_wmma<<<dim3((CA/16)*(TDE/32), BB), 32, 0, stream>>>(
        qw3h, q2h, qb3, qh, CA, CQP, TDE, 1, CA, 0, TDE, 0);

    // 3) squared norms of fp16 q/k (self-consistent with fp16 dot product)
    norms_kernel<<<gs((size_t)BB * TEN, 128) , 128, 0, stream>>>(kh, k2n, (size_t)BB * TEN, CA);
    norms_kernel<<<gs((size_t)BB * TDE, 128) , 128, 0, stream>>>(qh, q2n, (size_t)BB * TDE, CA);

    // 4) prior transpose (coalesced both ways via LDS tile)
    transpose_prior<<<dim3(TDE/32, TEN/32, BB), dim3(32, 8), 0, stream>>>(prior, priT);

    // 5) distance logits via WMMA GEMM, written into attn_logp region
    score_gemm_wmma<<<dim3((TDE/16)*(TEN/32), BB), 32, 0, stream>>>(qh, kh, q2n, k2n, alp);

    // 6) fused log_softmax + log-prior + softmax (one block per (b,t) row)
    softmax_kernel<<<(unsigned)(BB * TDE), 256, 0, stream>>>(alp, attn, priT);

    // 7) monotonic alignment search
    mas_fwd<<<BB, TEN, 0, stream>>>(attn, dirs);
    mas_bwd<<<BB, TEN, 0, stream>>>(dirs, mas, dur);
}
